// Encoder_45526653337951
// MI455X (gfx1250) — matrix operand order repaired and checked by reading, not compiled, not run on hardware
//
#include <hip/hip_runtime.h>

// ---------------------------------------------------------------------------
// Persistent LSTM-with-cell-LayerNorm encoder for MI455X (gfx1250, wave32).
//  - bf16 WMMA (v_wmma_f32_16x16x32_bf16), fp32 accumulate / pointwise / LN.
//  - K-fused recurrent GEMM: gates = [x_t ; h_{t-1}] @ [W_i;W_h]^T + (b_i+b_h)
//  - One workgroup per WGP (64 blocks): 192KB weight slice staged into LDS
//    ONCE in WMMA B-fragment order; re-read 512x from LDS, never from L2.
//  - Whole T=512 recurrence inside one kernel; device-scope atomic grid
//    barriers between GEMM phase and cell/LN phase. Cell state c stays in
//    registers of its owning block for the entire sequence.
// ---------------------------------------------------------------------------

typedef __attribute__((ext_vector_type(16))) __bf16 v16bf;
typedef __attribute__((ext_vector_type(8)))  __bf16 v8bf;
typedef __attribute__((ext_vector_type(8)))  float  v8f;

#define B_ 64
#define T_ 512
#define E_ 512
#define H_ 1024
#define G_ 4096   // 4*H
#define K_ 1536   // E + H

#define NBLK 64   // one block per N-tile of 64 AND per batch row
#define TPB  128  // 4 waves

// workspace layout (bytes)
#define OFF_WCT  0u            // WcT   bf16 [4096][1536]    12,582,912 B
#define OFF_BC   12582912u     // bc    f32  [4096]              16,384 B
#define OFF_EMB  12599296u     // emb   bf16 [64][512][512]  33,554,432 B
#define OFF_H    46153728u     // h     bf16 [64][1024]         131,072 B
#define OFF_G    46284800u     // gates f32  [64][4096]       1,048,576 B
#define OFF_BAR  47333376u     // barrier counter (u32)

#define LDSB_BYTES 196608u     // 64n x 1536k bf16, fragment-shuffled
#define DYN_LDS    (LDSB_BYTES + 2 * TPB * sizeof(float))

// --------------------------- prep kernels ----------------------------------

__global__ __launch_bounds__(256)
void enc_prep_w(const float* __restrict__ Wi, const float* __restrict__ Wh,
                __bf16* __restrict__ WcT)
{
    size_t idx = (size_t)blockIdx.x * 256 + threadIdx.x;
    if (idx >= (size_t)G_ * K_) return;
    int n = (int)(idx / K_);
    int k = (int)(idx % K_);
    float v = (k < E_) ? Wi[(size_t)k * G_ + n] : Wh[(size_t)(k - E_) * G_ + n];
    WcT[idx] = (__bf16)v;
}

__global__ __launch_bounds__(256)
void enc_prep_misc(const float* __restrict__ bi, const float* __restrict__ bh,
                   float* __restrict__ bc, __bf16* __restrict__ hBF,
                   unsigned* __restrict__ barCtr)
{
    int i = blockIdx.x * 256 + threadIdx.x;
    if (i < G_) bc[i] = bi[i] + bh[i];
    if (i < B_ * H_) hBF[i] = (__bf16)0.0f;
    if (i == 0) *barCtr = 0u;
}

__global__ __launch_bounds__(256)
void enc_embed(const int* __restrict__ src, const float* __restrict__ tab,
               __bf16* __restrict__ embBF)
{
    size_t idx = (size_t)blockIdx.x * 256 + threadIdx.x;   // over B*T*E
    int    e   = (int)(idx & (E_ - 1));
    size_t bt  = idx >> 9;                                  // E_ == 512
    int    tok = src[bt];
    float  v   = (tok != 0) ? tab[(size_t)tok * E_ + e] : 0.0f;
    embBF[idx] = (__bf16)v;
}

// --------------------------- fragment loaders ------------------------------
// A (16x32 bf16, ISA 7.12.2): lane m<16 holds K {0..7,16..23}, lane m+16 holds
// K {8..15,24..31} -> two 16B global loads with a 32B gap.
__device__ __forceinline__ v16bf ld_afrag(const __bf16* p) {
    v8bf lo = *(const v8bf*)(p);
    v8bf hi = *(const v8bf*)(p + 16);
    v16bf r;
#pragma unroll
    for (int i = 0; i < 8; ++i) { r[i] = lo[i]; r[i + 8] = hi[i]; }
    return r;
}
// B fragment from LDS, pre-shuffled so each (kkI,tile) fragment is two
// lane-contiguous 512B chunks: chunk c at elem ((kkI*8+tile*2... see staging).
__device__ __forceinline__ v16bf ld_bfragL(const __bf16* p) {
    v8bf lo = *(const v8bf*)(p);          // chunk 0: lane*8
    v8bf hi = *(const v8bf*)(p + 256);    // chunk 1: +32 lanes * 8 elems
    v16bf r;
#pragma unroll
    for (int i = 0; i < 8; ++i) { r[i] = lo[i]; r[i + 8] = hi[i]; }
    return r;
}

#define WMMA_BF16(A, Bf, C) \
    __builtin_amdgcn_wmma_f32_16x16x32_bf16(false, (A), false, (Bf), (short)0, (C), false, false)

// --------------------------- grid barrier ----------------------------------
__device__ __forceinline__ void gridbar(unsigned* ctr, unsigned target) {
    __threadfence();          // flush this thread's global stores (device scope)
    __syncthreads();          // all threads in block done + fenced
    if (threadIdx.x == 0) {
        atomicAdd(ctr, 1u);
        while (__hip_atomic_load(ctr, __ATOMIC_ACQUIRE, __HIP_MEMORY_SCOPE_AGENT) < target)
            __builtin_amdgcn_s_sleep(1);
    }
    __syncthreads();
    __threadfence();
}

// --------------------------- persistent kernel -----------------------------
__global__ __launch_bounds__(TPB)
void enc_persist(const __bf16* __restrict__ embBF, __bf16* __restrict__ hBF,
                 const __bf16* __restrict__ WcT, const float* __restrict__ bc,
                 float* __restrict__ gates, const float* __restrict__ gamma,
                 const float* __restrict__ beta, float* __restrict__ out,
                 unsigned* __restrict__ barCtr)
{
    extern __shared__ char smemRaw[];
    __bf16* ldsB = (__bf16*)smemRaw;                       // 192 KB weights
    float*  red  = (float*)(smemRaw + LDSB_BYTES);         // 2*TPB reduction

    const int tid  = threadIdx.x;
    const int bid  = blockIdx.x;          // N-tile owner AND batch-row owner
    const int lane = tid & 31;
    const int wave = tid >> 5;            // M tile 0..3
    const int c16  = lane & 15;
    const int half = lane >> 4;
    const int kbA  = half << 3;
    const int n0   = bid * 64;
    const int row  = wave * 16 + c16;     // batch row this lane loads for A

    // ---- stage weight slice into LDS in exact B-fragment order (once) ----
    // ci = kkI*256 + tile*64 + c*32 + l ; element = WcT[n0+tile*16+(l&15)]
    //                                              [kkI*32 + 8*(l>>4) + c*16 ..+8)
    for (int ci = tid; ci < (K_ / 32) * 4 * 2 * 32; ci += TPB) {
        const int l    = ci & 31;
        const int c    = (ci >> 5) & 1;
        const int tile = (ci >> 6) & 3;
        const int kkI  = ci >> 8;
        const int nl   = tile * 16 + (l & 15);
        const int kb   = kkI * 32 + ((l >> 4) << 3) + c * 16;
        *(v8bf*)(ldsB + (size_t)ci * 8) =
            *(const v8bf*)(WcT + (size_t)(n0 + nl) * K_ + kb);
    }

    // ---- loop-invariant register preloads ----
    const float bb0 = bc[n0 +  0 + c16];
    const float bb1 = bc[n0 + 16 + c16];
    const float bb2 = bc[n0 + 32 + c16];
    const float bb3 = bc[n0 + 48 + c16];
    float cg[8], cb[8], cReg[8];
#pragma unroll
    for (int j = 0; j < 8; ++j) {
        const int h = tid + j * TPB;
        cg[j] = gamma[h]; cb[j] = beta[h]; cReg[j] = 0.0f;  // c lives in regs
    }
    const __bf16* hrow = hBF + (size_t)row * H_ + kbA;
    const __bf16* lb   = ldsB + (size_t)lane * 8;
    const int     mB   = wave * 16 + (half << 3);

    __syncthreads();   // LDS weights ready

    unsigned bar = 0;
    for (int t = 0; t < T_; ++t) {
        // ============ GEMM phase: gates[64][4096] for this block's N range ==
        const __bf16* xrow = embBF + (((size_t)row * T_ + t) * E_) + kbA;
        v8f a0 = {}, a1 = {}, a2 = {}, a3 = {};
        int kkI = 0;
#pragma unroll 4
        for (int kk = 0; kk < E_; kk += 32, ++kkI) {           // x-part
            v16bf A = ld_afrag(xrow + kk);
            const __bf16* f = lb + (size_t)kkI * 2048;
            a0 = WMMA_BF16(A, ld_bfragL(f       ), a0);
            a1 = WMMA_BF16(A, ld_bfragL(f +  512), a1);
            a2 = WMMA_BF16(A, ld_bfragL(f + 1024), a2);
            a3 = WMMA_BF16(A, ld_bfragL(f + 1536), a3);
        }
#pragma unroll 4
        for (int kk = 0; kk < H_; kk += 32, ++kkI) {           // h-part
            v16bf A = ld_afrag(hrow + kk);
            const __bf16* f = lb + (size_t)kkI * 2048;
            a0 = WMMA_BF16(A, ld_bfragL(f       ), a0);
            a1 = WMMA_BF16(A, ld_bfragL(f +  512), a1);
            a2 = WMMA_BF16(A, ld_bfragL(f + 1024), a2);
            a3 = WMMA_BF16(A, ld_bfragL(f + 1536), a3);
        }
        // epilogue: C/D row = mtile*16 + 8*half + v, col = n0 + tile*16 + c16
        {
            const int col = n0 + 0 + c16;
#pragma unroll
            for (int v = 0; v < 8; ++v) gates[(size_t)(mB + v) * G_ + col] = a0[v] + bb0;
        }
        {
            const int col = n0 + 16 + c16;
#pragma unroll
            for (int v = 0; v < 8; ++v) gates[(size_t)(mB + v) * G_ + col] = a1[v] + bb1;
        }
        {
            const int col = n0 + 32 + c16;
#pragma unroll
            for (int v = 0; v < 8; ++v) gates[(size_t)(mB + v) * G_ + col] = a2[v] + bb2;
        }
        {
            const int col = n0 + 48 + c16;
#pragma unroll
            for (int v = 0; v < 8; ++v) gates[(size_t)(mB + v) * G_ + col] = a3[v] + bb3;
        }

        gridbar(barCtr, (++bar) * NBLK);   // all gates visible device-wide

        // ============ cell phase: this block owns batch row `bid` ===========
        float cv[8], ov[8], sum = 0.0f, sq = 0.0f;
#pragma unroll
        for (int j = 0; j < 8; ++j) {
            const int h = tid + j * TPB;
            float ig = gates[(size_t)bid * G_ + h];
            float fg = gates[(size_t)bid * G_ + H_ + h];
            float og = gates[(size_t)bid * G_ + 2 * H_ + h];
            float gg = gates[(size_t)bid * G_ + 3 * H_ + h];
            ig = 1.0f / (1.0f + __expf(-ig));
            fg = 1.0f / (1.0f + __expf(-fg));
            og = 1.0f / (1.0f + __expf(-og));
            const float cn = fg * cReg[j] + ig * tanhf(gg);
            cv[j] = cn; ov[j] = og;
            sum += cn; sq += cn * cn;
        }
        red[tid] = sum; red[TPB + tid] = sq;
        __syncthreads();
#pragma unroll
        for (int s = TPB / 2; s > 0; s >>= 1) {
            if (tid < s) { red[tid] += red[tid + s]; red[TPB + tid] += red[TPB + tid + s]; }
            __syncthreads();
        }
        const float mu  = red[0] * (1.0f / (float)H_);
        const float var = red[TPB] * (1.0f / (float)H_) - mu * mu;
        const float rs  = rsqrtf(var + 1e-5f);
#pragma unroll
        for (int j = 0; j < 8; ++j) {
            const int h  = tid + j * TPB;
            const float cn = (cv[j] - mu) * rs * cg[j] + cb[j];
            cReg[j] = cn;
            const float hv = ov[j] * tanhf(cn);
            hBF[(size_t)bid * H_ + h] = (__bf16)hv;
            out[((size_t)bid * T_ + t) * H_ + h] = hv;
            if (t == T_ - 1) {
                out[(size_t)B_ * T_ * H_ + (size_t)bid * H_ + h] = hv;                    // h_f
                out[(size_t)B_ * T_ * H_ + (size_t)B_ * H_ + (size_t)bid * H_ + h] = cn;  // c_f
            }
        }

        gridbar(barCtr, (++bar) * NBLK);   // new h visible before next GEMM
    }
}

// ------------------------------- launcher ----------------------------------
extern "C" void kernel_launch(void* const* d_in, const int* in_sizes, int n_in,
                              void* d_out, int out_size, void* d_ws, size_t ws_size,
                              hipStream_t stream)
{
    const int*   src = (const int*)  d_in[0];
    const float* tab = (const float*)d_in[1];
    const float* Wi  = (const float*)d_in[2];
    const float* bi  = (const float*)d_in[3];
    const float* Wh  = (const float*)d_in[4];
    const float* bh  = (const float*)d_in[5];
    const float* gam = (const float*)d_in[6];
    const float* bet = (const float*)d_in[7];
    float* out = (float*)d_out;

    char* ws = (char*)d_ws;
    __bf16*   WcT = (__bf16*)  (ws + OFF_WCT);
    float*    bc  = (float*)   (ws + OFF_BC);
    __bf16*   emb = (__bf16*)  (ws + OFF_EMB);
    __bf16*   hBF = (__bf16*)  (ws + OFF_H);
    float*    g   = (float*)   (ws + OFF_G);
    unsigned* bar = (unsigned*)(ws + OFF_BAR);

    enc_prep_w   <<<(unsigned)(((size_t)G_ * K_ + 255) / 256), 256, 0, stream>>>(Wi, Wh, WcT);
    enc_prep_misc<<<(B_ * H_ + 255) / 256, 256, 0, stream>>>(bi, bh, bc, hBF, bar);
    enc_embed    <<<(unsigned)(((size_t)B_ * T_ * E_ + 255) / 256), 256, 0, stream>>>(src, tab, emb);

    enc_persist<<<NBLK, TPB, DYN_LDS, stream>>>(emb, hBF, WcT, bc, g, gam, bet, out, bar);
}
